// Attention_71227737637519
// MI455X (gfx1250) — matrix operand order repaired and checked by reading, not compiled, not run on hardware
//
#include <hip/hip_runtime.h>
#include <hip/hip_bf16.h>

// ---------------------------------------------------------------------------
// MI455X (gfx1250) attention layer:
//   Q/K/V proj (bf16 WMMA, f32 accum) -> RoPE -> flash attention (WMMA) ->
//   output proj (bf16 WMMA -> f32).
// All matrix math on v_wmma_f32_16x16x32_bf16 (wave32, 16x16x32).
// V is staged TRANSPOSED in LDS so every WMMA operand is a contiguous
// ds_load_b128 pair; f32->bf16 uses packed v_cvt_pk_bf16_f32.
// ---------------------------------------------------------------------------

#define N_HEADS    32
#define N_KV_HEADS 8
#define HEAD_DIM   128
#define DIM        4096
#define BSZ        2
#define SEQLEN     2048
#define M_ROWS     (BSZ * SEQLEN)   // 4096
#define KVD        (N_KV_HEADS * HEAD_DIM)  // 1024

typedef __attribute__((ext_vector_type(16))) __bf16 v16bf;
typedef __attribute__((ext_vector_type(8)))  float  v8f;

union Frag {
    v16bf v;
    uint4 q[2];
    unsigned short h[16];
};

__device__ __forceinline__ unsigned short f2bf(float f) {
    union { float f; unsigned u; } x{f};
    unsigned r = x.u + 0x7FFFu + ((x.u >> 16) & 1u);   // round-to-nearest-even
    return (unsigned short)(r >> 16);
}
__device__ __forceinline__ float bf2f(unsigned short b) {
    union { unsigned u; float f; } x{(unsigned)b << 16};
    return x.f;
}
// packed f32x2 -> bf16x2 (v_cvt_pk_bf16_f32)
__device__ __forceinline__ unsigned pack2bf(float x, float y) {
    float2 f; f.x = x; f.y = y;
    __hip_bfloat162 h = __float22bfloat162_rn(f);
    unsigned u; __builtin_memcpy(&u, &h, 4);
    return u;
}

// ---------------------------------------------------------------------------
// Generic GEMM: C[M,N] = A[M,K] * W[N,K]^T
//   A: f32 (A_BF16=false) or bf16 (A_BF16=true), row-major
//   W: f32 row-major [N][K] (converted to bf16 on load)
//   C: bf16 (OUT_BF16=true) or f32, row-major
// Block tile 64x64, BK=64, 256 threads = 8 waves, wave -> 16x32 C patch,
// 4 WMMA per wave per barrier pair.
// ---------------------------------------------------------------------------
template <bool A_BF16, bool OUT_BF16>
__global__ __launch_bounds__(256) void gemm_xwT_kernel(
    const void* __restrict__ Aptr, const float* __restrict__ W,
    void* __restrict__ Cptr, int M, int N, int K)
{
    constexpr int BM = 64, BN = 64, BK = 64, LDSR = 72;  // 72-u16 stride (pad)
    __shared__ __align__(16) unsigned short Asm[BM * LDSR];
    __shared__ __align__(16) unsigned short Bsm[BN * LDSR];

    const int tid  = threadIdx.x;
    const int w    = tid >> 5;
    const int lane = tid & 31;
    const int r    = lane & 15;       // WMMA row/col index within 16
    const int hi   = lane >> 4;       // lane-half selector
    const int wr   = w >> 1;          // 0..3  -> 16-row slab
    const int wc   = w & 1;           // 0..1  -> 32-col slab
    const int m0   = blockIdx.y * BM;
    const int n0   = blockIdx.x * BN;

    v8f acc[2];
#pragma unroll
    for (int t = 0; t < 2; ++t)
#pragma unroll
        for (int i = 0; i < 8; ++i) acc[t][i] = 0.0f;

    const float*          Af = (const float*)Aptr;
    const unsigned short* Ab = (const unsigned short*)Aptr;

    for (int kpos = 0; kpos < K; kpos += BK) {
        __syncthreads();
        // ---- stage A tile (64 x 64) into LDS as bf16 ----
        if constexpr (A_BF16) {
#pragma unroll
            for (int it = 0; it < 2; ++it) {
                const int e = tid + it * 256;
                const int row = e >> 3, c8 = e & 7;       // 8x(8 bf16) per row
                *(uint4*)&Asm[row * LDSR + c8 * 8] =
                    *(const uint4*)(Ab + (size_t)(m0 + row) * K + kpos + c8 * 8);
            }
        } else {
#pragma unroll
            for (int it = 0; it < 4; ++it) {
                const int e = tid + it * 256;
                const int row = e >> 4, c4 = e & 15;      // 16x(4 f32) per row
                const float4 v =
                    *(const float4*)(Af + (size_t)(m0 + row) * K + kpos + c4 * 4);
                uint2 p;
                p.x = pack2bf(v.x, v.y);
                p.y = pack2bf(v.z, v.w);
                *(uint2*)&Asm[row * LDSR + c4 * 4] = p;
            }
        }
        // ---- stage W tile (64 x 64) into LDS as bf16 ----
#pragma unroll
        for (int it = 0; it < 4; ++it) {
            const int e = tid + it * 256;
            const int row = e >> 4, c4 = e & 15;
            const float* src = W + (size_t)(n0 + row) * K + kpos + c4 * 4;
            if (kpos + BK < K)
                __builtin_prefetch(src + BK, 0, 0);       // global_prefetch_b8
            const float4 v = *(const float4*)src;
            uint2 p;
            p.x = pack2bf(v.x, v.y);
            p.y = pack2bf(v.z, v.w);
            *(uint2*)&Bsm[row * LDSR + c4 * 4] = p;
        }
        __syncthreads();

#pragma unroll
        for (int kk = 0; kk < BK; kk += 32) {
            // A fragment: lane = row r, K-chunks [hi*8..+7],[16+hi*8..+7]
            Frag a;
            const unsigned short* arow = &Asm[(wr * 16 + r) * LDSR + kk];
            a.q[0] = *(const uint4*)(arow + hi * 8);
            a.q[1] = *(const uint4*)(arow + 16 + hi * 8);
#pragma unroll
            for (int t = 0; t < 2; ++t) {
                // B fragment: lane = col r, K-chunks [hi*8..+7],[16+hi*8..+7]
                Frag bfr;
                const unsigned short* brow =
                    &Bsm[(wc * 32 + t * 16 + r) * LDSR + kk];
                bfr.q[0] = *(const uint4*)(brow + hi * 8);
                bfr.q[1] = *(const uint4*)(brow + 16 + hi * 8);
                acc[t] = __builtin_amdgcn_wmma_f32_16x16x32_bf16(
                    false, a.v, false, bfr.v, (short)0, acc[t], false, false);
            }
        }
    }

    // ---- epilogue: C frag -> row = hi*8+i, col = lane&15 ----
#pragma unroll
    for (int t = 0; t < 2; ++t) {
        const int col = n0 + wc * 32 + t * 16 + r;
#pragma unroll
        for (int i = 0; i < 8; ++i) {
            const int rowg = m0 + wr * 16 + hi * 8 + i;
            if constexpr (OUT_BF16)
                ((unsigned short*)Cptr)[(size_t)rowg * N + col] = f2bf(acc[t][i]);
            else
                ((float*)Cptr)[(size_t)rowg * N + col] = acc[t][i];
        }
    }
}

// ---------------------------------------------------------------------------
// RoPE on a bf16 [BSZ*SEQLEN][nheads*HEAD_DIM] buffer; pairs (2p, 2p+1).
// ---------------------------------------------------------------------------
__global__ void rope_kernel(unsigned short* __restrict__ buf,
                            const float* __restrict__ fcos,
                            const float* __restrict__ fsin,
                            int nheads, long total)
{
    long idx = (long)blockIdx.x * blockDim.x + threadIdx.x;
    if (idx >= total) return;
    const int  p   = (int)(idx & 63);          // pair index, HEAD_DIM/2 = 64
    long       tmp = idx >> 6;
    const int  hh  = (int)(tmp % nheads);
    const long row = tmp / nheads;             // b*SEQLEN + s
    const int  s   = (int)(row % SEQLEN);
    const float c  = fcos[s * 64 + p];
    const float sn = fsin[s * 64 + p];
    unsigned short* e = buf + row * ((long)nheads * HEAD_DIM) + hh * HEAD_DIM + p * 2;
    const float a = bf2f(e[0]);
    const float b = bf2f(e[1]);
    *(unsigned*)e = pack2bf(a * c - b * sn, a * sn + b * c);
}

// ---------------------------------------------------------------------------
// Flash attention: block = (b, h, 64 q-rows); 4 waves x 16 q-rows.
// K staged row-major [key][dim]; V staged TRANSPOSED [dim][key] so the
// P*V B-fragment is two contiguous ds_load_b128. Online softmax.
// ---------------------------------------------------------------------------
__global__ __launch_bounds__(256) void flash_attn_kernel(
    const unsigned short* __restrict__ Q,    // [M_ROWS][DIM] bf16
    const unsigned short* __restrict__ Kc,   // [M_ROWS][KVD] bf16
    const unsigned short* __restrict__ Vc,   // [M_ROWS][KVD] bf16
    unsigned short* __restrict__ Oa)         // [M_ROWS][DIM] bf16
{
    constexpr int KSTR  = 136;               // 128 + 8 u16 pad (K tile row)
    constexpr int VSTRT = 40;                // 32 + 8 u16 pad (V^T tile row)
    constexpr int PSTR  = 40;                // 32 + 8 u16 pad (P tile row)
    __shared__ __align__(16) unsigned short Ksm[32 * KSTR];        // [key][dim]
    __shared__ __align__(16) unsigned short Vsmt[HEAD_DIM * VSTRT];// [dim][key]
    __shared__ __align__(16) unsigned short Psm[4 * 16 * PSTR];

    const int qb = blockIdx.x, h = blockIdx.y, b = blockIdx.z;
    const int kvh  = h >> 2;                 // n_rep = 32/8 = 4
    const int tid  = threadIdx.x;
    const int w    = tid >> 5;
    const int lane = tid & 31;
    const int r    = lane & 15;
    const int hi   = lane >> 4;
    const int q0   = qb * 64;
    const int qw   = q0 + w * 16;            // this wave's first q row

    // ---- Q fragments (held in registers for entire KV loop) ----
    Frag qf[4];
    const unsigned short* qbase =
        Q + (size_t)(b * SEQLEN + qw + r) * DIM + h * HEAD_DIM;
#pragma unroll
    for (int kk = 0; kk < 4; ++kk) {
        qf[kk].q[0] = *(const uint4*)(qbase + kk * 32 + hi * 8);
        qf[kk].q[1] = *(const uint4*)(qbase + kk * 32 + 16 + hi * 8);
    }

    v8f o[8];
#pragma unroll
    for (int nt = 0; nt < 8; ++nt)
#pragma unroll
        for (int i = 0; i < 8; ++i) o[nt][i] = 0.0f;
    float mrow[8], lrow[8];
#pragma unroll
    for (int i = 0; i < 8; ++i) { mrow[i] = -1e30f; lrow[i] = 0.0f; }

    const int nkb  = (q0 + 64) >> 5;         // causal: keys 0 .. q0+63
    const int krow = tid >> 3, kc = tid & 7; // K staging: 32 rows x 8 chunks x2
    const int kp   = tid >> 4, ch = tid & 15;// V staging: 16 key-pairs x 16 chunks
    const float scale = 0.08838834764831845f; // 1/sqrt(128)

    for (int kb = 0; kb < nkb; ++kb) {
        const int kv0 = kb * 32;
        __syncthreads();
        {   // ---- stage K rows (row-major) ----
            const size_t gk =
                (size_t)(b * SEQLEN + kv0 + krow) * KVD + kvh * HEAD_DIM + kc * 16;
            const uint4* sk = (const uint4*)(Kc + gk);
            *(uint4*)&Ksm[krow * KSTR + kc * 16]     = sk[0];
            *(uint4*)&Ksm[krow * KSTR + kc * 16 + 8] = sk[1];
            // ---- stage V transposed: keys 2kp,2kp+1 / dims ch*8..+7 ----
            const size_t gv =
                (size_t)(b * SEQLEN + kv0 + 2 * kp) * KVD + kvh * HEAD_DIM + ch * 8;
            const uint4 a0 = *(const uint4*)(Vc + gv);
            const uint4 a1 = *(const uint4*)(Vc + gv + KVD);
            const unsigned short* s0 = (const unsigned short*)&a0;
            const unsigned short* s1 = (const unsigned short*)&a1;
#pragma unroll
            for (int d = 0; d < 8; ++d) {
                const unsigned pv = (unsigned)s0[d] | ((unsigned)s1[d] << 16);
                *(unsigned*)&Vsmt[(ch * 8 + d) * VSTRT + 2 * kp] = pv;
            }
        }
        __syncthreads();

        // ---- scores S = Q * K^T : 2 key tiles, K-dim 128 = 4 x 32 ----
        v8f sc[2];
#pragma unroll
        for (int t = 0; t < 2; ++t)
#pragma unroll
            for (int i = 0; i < 8; ++i) sc[t][i] = 0.0f;
#pragma unroll
        for (int kk = 0; kk < 4; ++kk) {
#pragma unroll
            for (int t = 0; t < 2; ++t) {
                Frag bfr;
                const unsigned short* kr_ =
                    &Ksm[(t * 16 + r) * KSTR + kk * 32];
                bfr.q[0] = *(const uint4*)(kr_ + hi * 8);
                bfr.q[1] = *(const uint4*)(kr_ + 16 + hi * 8);
                sc[t] = __builtin_amdgcn_wmma_f32_16x16x32_bf16(
                    false, qf[kk].v, false, bfr.v, (short)0, sc[t], false, false);
            }
        }

        // ---- online softmax (row = hi*8+i, col = lane&15), fully unrolled ----
#pragma unroll
        for (int i = 0; i < 8; ++i) {
            const int qrow = qw + hi * 8 + i;
            float s0 = sc[0][i] * scale;
            float s1 = sc[1][i] * scale;
            if (kv0 + r      > qrow) s0 = -1e30f;   // causal mask
            if (kv0 + 16 + r > qrow) s1 = -1e30f;
            float t = fmaxf(s0, s1);
#pragma unroll
            for (int off = 1; off < 16; off <<= 1)
                t = fmaxf(t, __shfl_xor(t, off, 32));
            const float mn    = fmaxf(mrow[i], t);
            const float alpha = __expf(mrow[i] - mn);
            const float p0    = __expf(s0 - mn);
            const float p1    = __expf(s1 - mn);
            float ps = p0 + p1;
#pragma unroll
            for (int off = 1; off < 16; off <<= 1)
                ps += __shfl_xor(ps, off, 32);
            lrow[i] = lrow[i] * alpha + ps;
            mrow[i] = mn;
#pragma unroll
            for (int nt = 0; nt < 8; ++nt) o[nt][i] *= alpha;
            // P (probabilities) -> wave-private LDS, C-layout scatter
            unsigned short* pr = &Psm[(w * 16 + hi * 8 + i) * PSTR];
            pr[r]      = f2bf(p0);
            pr[16 + r] = f2bf(p1);
        }

        // ---- O += P * V : both fragments are contiguous b128 pairs ----
        Frag pa;
        const unsigned short* prow = &Psm[(w * 16 + r) * PSTR];
        pa.q[0] = *(const uint4*)(prow + hi * 8);
        pa.q[1] = *(const uint4*)(prow + 16 + hi * 8);
#pragma unroll
        for (int nt = 0; nt < 8; ++nt) {
            Frag vb;
            const unsigned short* vrow = &Vsmt[(nt * 16 + r) * VSTRT];
            vb.q[0] = *(const uint4*)(vrow + hi * 8);
            vb.q[1] = *(const uint4*)(vrow + 16 + hi * 8);
            o[nt] = __builtin_amdgcn_wmma_f32_16x16x32_bf16(
                false, pa.v, false, vb.v, (short)0, o[nt], false, false);
        }
    }

    // ---- normalize and store ----
#pragma unroll
    for (int i = 0; i < 8; ++i) {
        const float inv = 1.0f / lrow[i];
        const size_t rowg =
            (size_t)(b * SEQLEN + qw + hi * 8 + i) * DIM + h * HEAD_DIM;
#pragma unroll
        for (int nt = 0; nt < 8; ++nt)
            Oa[rowg + nt * 16 + r] = f2bf(o[nt][i] * inv);
    }
}

// ---------------------------------------------------------------------------
// Launch: QKV proj -> RoPE(Q), RoPE(K) -> flash attention -> O proj.
// ---------------------------------------------------------------------------
extern "C" void kernel_launch(void* const* d_in, const int* in_sizes, int n_in,
                              void* d_out, int out_size, void* d_ws, size_t ws_size,
                              hipStream_t stream)
{
    (void)in_sizes; (void)n_in; (void)out_size; (void)ws_size;
    const float* x    = (const float*)d_in[0];
    const float* wq   = (const float*)d_in[1];
    const float* wk   = (const float*)d_in[2];
    const float* wv   = (const float*)d_in[3];
    const float* wo   = (const float*)d_in[4];
    const float* fcos = (const float*)d_in[7];
    const float* fsin = (const float*)d_in[8];
    float* out = (float*)d_out;

    // workspace layout (bf16): Q[4096x4096] K[4096x1024] V[4096x1024] A[4096x4096]
    unsigned short* qbuf = (unsigned short*)d_ws;
    unsigned short* kbuf = qbuf + (size_t)M_ROWS * DIM;
    unsigned short* vbuf = kbuf + (size_t)M_ROWS * KVD;
    unsigned short* abuf = vbuf + (size_t)M_ROWS * KVD;

    const dim3 blk(256);
    gemm_xwT_kernel<false, true><<<dim3(DIM / 64, M_ROWS / 64), blk, 0, stream>>>(
        x, wq, qbuf, M_ROWS, DIM, DIM);
    gemm_xwT_kernel<false, true><<<dim3(KVD / 64, M_ROWS / 64), blk, 0, stream>>>(
        x, wk, kbuf, M_ROWS, KVD, DIM);
    gemm_xwT_kernel<false, true><<<dim3(KVD / 64, M_ROWS / 64), blk, 0, stream>>>(
        x, wv, vbuf, M_ROWS, KVD, DIM);

    const long tq = (long)M_ROWS * N_HEADS * (HEAD_DIM / 2);
    rope_kernel<<<(unsigned)((tq + 255) / 256), blk, 0, stream>>>(qbuf, fcos, fsin, N_HEADS, tq);
    const long tk = (long)M_ROWS * N_KV_HEADS * (HEAD_DIM / 2);
    rope_kernel<<<(unsigned)((tk + 255) / 256), blk, 0, stream>>>(kbuf, fcos, fsin, N_KV_HEADS, tk);

    flash_attn_kernel<<<dim3(SEQLEN / 64, N_HEADS, BSZ), blk, 0, stream>>>(
        qbuf, kbuf, vbuf, abuf);

    gemm_xwT_kernel<true, false><<<dim3(DIM / 64, M_ROWS / 64), blk, 0, stream>>>(
        abuf, wo, out, M_ROWS, DIM, DIM);
}
